// HopfieldDQN_88218628260372
// MI455X (gfx1250) — matrix operand order repaired and checked by reading, not compiled, not run on hardware
//
#include <hip/hip_runtime.h>

typedef __attribute__((ext_vector_type(16))) _Float16 v16h;
typedef __attribute__((ext_vector_type(8)))  _Float16 v8h;
typedef __attribute__((ext_vector_type(8)))  float    v8f;
typedef __attribute__((ext_vector_type(4)))  unsigned v4u;
typedef __attribute__((ext_vector_type(8)))  unsigned v8u;

#define USE_TDM 1

#define BATCH 65536
#define IN    256
#define HID   512
#define EP    18
#define OUTD  18
#define CIN   274     // IN + EP
#define NMEM  100
#define TM    64      // batch rows per block
#define NTHREADS 256  // 8 wave32s

// ---- workspace byte offsets (all 256-aligned) ----
#define OFF_W1   0u         // f16 [512][256]
#define OFF_W3X  262144u    // f16 [512][256]
#define OFF_W3R  524288u    // f16 [512][32]
#define OFF_W4   557056u    // f16 [512][512]
#define OFF_W2P  1081344u   // f16 [32][512] (rows >=18 zero)
#define OFF_W5P  1114112u   // f16 [32][512] (rows >=18 zero)
#define OFF_MM   1146880u   // u32 [100] sign masks

// ---- LDS layout (bytes) ----
#define XA_STRIDE 272       // halfs per row of x tile  [64][272]
#define H_STRIDE  528       // halfs per row of h tiles [64][528]
#define HA_OFF    67584     // region0 [0,67584): xA then hB; staging f32 x goes in hA region
#define RBUF_OFF  (HA_OFF + 67584)         // f16 [64][32] retrieved
#define PM_OFF    (RBUF_OFF + 4096)        // u32 [64] probe masks
#define RM_OFF    (PM_OFF + 256)           // u32 [64] retrieved masks
#define MM_OFF    (RM_OFF + 256)           // u32 [128] memory masks
#define SMEM_B    (MM_OFF + 512)           // = 140288 bytes

// ---------------- prep: f32 -> f16 weights, padding, memory bitmasks ----------------
__global__ void hop_prep(const float* __restrict__ W1, const float* __restrict__ W2,
                         const float* __restrict__ W3, const float* __restrict__ W4,
                         const float* __restrict__ W5, const float* __restrict__ mem,
                         char* __restrict__ ws) {
    _Float16* w1h  = (_Float16*)(ws + OFF_W1);
    _Float16* w3xh = (_Float16*)(ws + OFF_W3X);
    _Float16* w3rh = (_Float16*)(ws + OFF_W3R);
    _Float16* w4h  = (_Float16*)(ws + OFF_W4);
    _Float16* w2p  = (_Float16*)(ws + OFF_W2P);
    _Float16* w5p  = (_Float16*)(ws + OFF_W5P);
    unsigned* mm   = (unsigned*)(ws + OFF_MM);

    int tid = blockIdx.x * blockDim.x + threadIdx.x;
    int nth = gridDim.x * blockDim.x;

    for (int i = tid; i < HID * IN; i += nth) w1h[i] = (_Float16)W1[i];
    for (int i = tid; i < HID * HID; i += nth) w4h[i] = (_Float16)W4[i];
    for (int i = tid; i < HID * IN; i += nth) {
        int r = i >> 8, c = i & 255;
        w3xh[i] = (_Float16)W3[r * CIN + c];
    }
    for (int i = tid; i < HID * 32; i += nth) {
        int r = i >> 5, c = i & 31;
        w3rh[i] = (c < EP) ? (_Float16)W3[r * CIN + IN + c] : (_Float16)0.0f;
    }
    for (int i = tid; i < 32 * HID; i += nth) {
        int r = i >> 9, c = i & 511;
        w2p[i] = (r < EP)   ? (_Float16)W2[r * HID + c] : (_Float16)0.0f;
    }
    for (int i = tid; i < 32 * HID; i += nth) {
        int r = i >> 9, c = i & 511;
        w5p[i] = (r < OUTD) ? (_Float16)W5[r * HID + c] : (_Float16)0.0f;
    }
    for (int j = tid; j < NMEM; j += nth) {
        unsigned m = 0u;
        for (int e = 0; e < EP; ++e)
            if (mem[j * EP + e] > 0.0f) m |= (1u << e);
        mm[j] = m;
    }
}

// ---------------- fragment loaders (layouts per CDNA5 ISA 7.12.2) ----------------
// A 16x32 f16: lane<16 holds row m=lane, K {k0..k0+7, k0+16..k0+23};
//              lane>=16 holds row m=lane-16, K {k0+8..k0+15, k0+24..k0+31}
__device__ __forceinline__ v16h load_afrag(const _Float16* base, int stride,
                                           int m0, int k0, int lane) {
    const int lh  = (lane < 16);
    const int row = m0 + (lane & 15);
    const _Float16* p = base + row * stride + k0;
    union { v16h v; v8h h[2]; } a;
    a.h[0] = *(const v8h*)(p + (lh ? 0 : 8));
    a.h[1] = *(const v8h*)(p + (lh ? 16 : 24));
    return a.v;
}
// B 32x16 f16 from row-major weight W[N][K]: column n of B == row n of W,
// lanes<16 take K {k0..k0+7, k0+16..k0+23}, lanes>=16 take K {k0+8..k0+15, k0+24..k0+31}.
__device__ __forceinline__ v16h load_bfrag(const _Float16* W, int ldb,
                                           int n0, int k0, int lane) {
    const int lh = (lane < 16);
    const int n  = n0 + (lane & 15);
    const _Float16* p = W + n * ldb + k0;
    union { v16h v; v8h h[2]; } b;
    b.h[0] = *(const v8h*)(p + (lh ? 0 : 8));
    b.h[1] = *(const v8h*)(p + (lh ? 16 : 24));
    return b.v;
}

__device__ __forceinline__ v8f wmma16(v16h a, v16h b, v8f c) {
    return __builtin_amdgcn_wmma_f32_16x16x32_f16(false, a, false, b,
                                                  (short)0, c, false, false);
}

__device__ __forceinline__ void store_relu(_Float16* h, int stride, int m0, int n,
                                           const v8f& c, float bias, int lh) {
    for (int r = 0; r < 8; ++r) {
        float v = c[r] + bias; v = v > 0.0f ? v : 0.0f;
        h[(m0 + r + (lh ? 0 : 8)) * stride + n] = (_Float16)v;
    }
}

// ---------------- fused forward kernel ----------------
__global__ __launch_bounds__(NTHREADS)
void hop_main(const float* __restrict__ x,
              const float* __restrict__ b1, const float* __restrict__ b2,
              const float* __restrict__ b3, const float* __restrict__ b4,
              const float* __restrict__ b5,
              const char* __restrict__ ws, float* __restrict__ out) {
    __shared__ __align__(16) char smem[SMEM_B];
    _Float16* xA   = (_Float16*)(smem);             // phase A: x tile f16 [64][272]
    _Float16* hB   = (_Float16*)(smem);             // phase B: h2 [64][528] (aliases xA)
    _Float16* hA   = (_Float16*)(smem + HA_OFF);    // [64][528]; also f32 x staging
    _Float16* rbuf = (_Float16*)(smem + RBUF_OFF);  // [64][32]
    unsigned* pm   = (unsigned*)(smem + PM_OFF);
    unsigned* rm   = (unsigned*)(smem + RM_OFF);
    unsigned* mmL  = (unsigned*)(smem + MM_OFF);

    const _Float16* W1h  = (const _Float16*)(ws + OFF_W1);
    const _Float16* W3xh = (const _Float16*)(ws + OFF_W3X);
    const _Float16* W3rh = (const _Float16*)(ws + OFF_W3R);
    const _Float16* W4h  = (const _Float16*)(ws + OFF_W4);
    const _Float16* W2p  = (const _Float16*)(ws + OFF_W2P);
    const _Float16* W5p  = (const _Float16*)(ws + OFF_W5P);
    const unsigned* mmG  = (const unsigned*)(ws + OFF_MM);

    const int tid  = threadIdx.x;
    const int wid  = tid >> 5;
    const int lane = tid & 31;
    const int lh   = (lane < 16);
    const int row0 = blockIdx.x * TM;

    // ---- stage 0: x tile [64][256] f32 -> LDS (TDM) -> f16 xA; memory masks -> LDS ----
#if USE_TDM
    if (wid == 0) {
        // Tensor DMA descriptor (D#), 2-group form: 1-D tile of 16384 f32 elements.
        unsigned long long ga = (unsigned long long)(uintptr_t)(x + (size_t)row0 * IN);
        unsigned ldsa = (unsigned)(uintptr_t)(void*)(smem + HA_OFF);
        v4u g0; v8u g1;
        g0[0] = 1u;                                  // count=1, user mode
        g0[1] = ldsa;                                // lds_addr
        g0[2] = (unsigned)ga;                        // global_addr[31:0]
        g0[3] = (unsigned)((ga >> 32) & 0x1FFFFFFu)  // global_addr[56:32]
              | (2u << 30);                          // type=2 (image)
        g1[0] = (2u << 16);                          // data_size=2 (4 bytes)
        g1[1] = (16384u << 16);                      // tensor_dim0[15:0]=16384
        g1[2] = (1u << 16);                          // tensor_dim0 hi=0 | tensor_dim1=1
        g1[3] = (16384u << 16);                      // tensor_dim1 hi=0 | tile_dim0=16384
        g1[4] = 0u;                                  // tile_dim1=0, tile_dim2=0 (1-D)
        g1[5] = 16384u;                              // tensor_dim0_stride lo
        g1[6] = 0u;
        g1[7] = 0u;
        asm volatile("tensor_load_to_lds %0, %1" :: "s"(g0), "s"(g1) : "memory");
        __builtin_amdgcn_s_wait_tensorcnt(0);
    }
    for (int i = tid; i < NMEM; i += NTHREADS) mmL[i] = mmG[i];
    __syncthreads();
    {   // convert staged f32 -> f16 with padded rows
        const float4* st = (const float4*)(smem + HA_OFF);
        for (int i = tid; i < TM * IN / 4; i += NTHREADS) {
            int r = i >> 6, c4 = i & 63;
            float4 v = st[i];
            _Float16* d = xA + r * XA_STRIDE + c4 * 4;
            d[0] = (_Float16)v.x; d[1] = (_Float16)v.y;
            d[2] = (_Float16)v.z; d[3] = (_Float16)v.w;
        }
    }
#else
    {
        const float4* xg = (const float4*)(x + (size_t)row0 * IN);
        for (int i = tid; i < TM * IN / 4; i += NTHREADS) {
            int r = i >> 6, c4 = i & 63;
            float4 v = xg[(size_t)r * 64 + c4];
            _Float16* d = xA + r * XA_STRIDE + c4 * 4;
            d[0] = (_Float16)v.x; d[1] = (_Float16)v.y;
            d[2] = (_Float16)v.z; d[3] = (_Float16)v.w;
        }
        for (int i = tid; i < NMEM; i += NTHREADS) mmL[i] = mmG[i];
    }
#endif
    __syncthreads();

    // ---- GEMM1: enc_h = relu(x @ W1^T + b1) -> hA [64][512] f16 ----
    // Each wave: 4 N-tiles x all 4 M-tiles; one B fragment feeds 4 WMMAs.
    for (int j = 0; j < 4; ++j) {
        const int n0 = (wid * 4 + j) * 16;
        v8f c0 = {}, c1 = {}, c2 = {}, c3 = {};
        for (int k0 = 0; k0 < IN; k0 += 32) {
            v16h b  = load_bfrag(W1h, IN, n0, k0, lane);
            v16h a0 = load_afrag(xA, XA_STRIDE, 0,  k0, lane);
            v16h a1 = load_afrag(xA, XA_STRIDE, 16, k0, lane);
            v16h a2 = load_afrag(xA, XA_STRIDE, 32, k0, lane);
            v16h a3 = load_afrag(xA, XA_STRIDE, 48, k0, lane);
            c0 = wmma16(a0, b, c0); c1 = wmma16(a1, b, c1);
            c2 = wmma16(a2, b, c2); c3 = wmma16(a3, b, c3);
        }
        const int n = n0 + (lane & 15);
        const float bias = b1[n];
        store_relu(hA, H_STRIDE, 0,  n, c0, bias, lh);
        store_relu(hA, H_STRIDE, 16, n, c1, bias, lh);
        store_relu(hA, H_STRIDE, 32, n, c2, bias, lh);
        store_relu(hA, H_STRIDE, 48, n, c3, bias, lh);
    }
    __syncthreads();

    // ---- GEMM2: enc = enc_h @ W2p^T + b2 ; sign -> probe masks via ballot ----
    {
        const int mt = wid & 3;         // one 16-row M-tile per wave here
        const int m0 = mt * 16;
        const int nh = wid >> 2;        // N-tile 0 (cols 0-15) or 1 (cols 16-31)
        const int n0 = nh * 16;
        v8f c = {};
        for (int k0 = 0; k0 < HID; k0 += 32) {
            v16h a = load_afrag(hA, H_STRIDE, m0, k0, lane);
            v16h b = load_bfrag(W2p, HID, n0, k0, lane);
            c = wmma16(a, b, c);
        }
        const int n = n0 + (lane & 15);
        const float bias = (n < EP) ? b2[n] : 0.0f;
        if (nh == 0) {  // cols 0..15 -> low 16 bits of probe mask
            for (int r = 0; r < 8; ++r) {
                unsigned bal = __builtin_amdgcn_ballot_w32((c[r] + bias) > 0.0f);
                if (lane == 0) {
                    pm[m0 + r]     = bal & 0xFFFFu;
                    pm[m0 + r + 8] = bal >> 16;
                }
            }
        }
        __syncthreads();
        if (nh == 1) {  // cols 16..17 -> bits 16..17
            for (int r = 0; r < 8; ++r) {
                unsigned bal = __builtin_amdgcn_ballot_w32((c[r] + bias) > 0.0f);
                if (lane == 0) {
                    pm[m0 + r]     |= (bal & 3u) << 16;
                    pm[m0 + r + 8] |= ((bal >> 16) & 3u) << 16;
                }
            }
        }
    }
    __syncthreads();

    // ---- Hopfield: argmax_j probe . mem_j  (sim = 18 - 2*popc(xor)) ----
    if (tid < TM) {
        unsigned mask = pm[tid] & 0x3FFFFu;
        int best = -1000, bi = 0;
        for (int j = 0; j < NMEM; ++j) {
            int d   = __builtin_popcount((mask ^ mmL[j]) & 0x3FFFFu);
            int sim = EP - 2 * d;
            if (sim > best) { best = sim; bi = j; }   // first-max tie rule
        }
        rm[tid] = mmL[bi];
    }
    __syncthreads();
    for (int i = tid; i < TM * 32; i += NTHREADS) {
        int r = i >> 5, cx = i & 31;
        float v = (cx < EP) ? (float)((rm[r] >> cx) & 1u) : 0.0f;
        rbuf[i] = (_Float16)v;
    }
    __syncthreads();

    // ---- GEMM3: h = relu(x@W3x^T + retrieved@W3r^T + b3) -> hA (enc_h dead) ----
    for (int j = 0; j < 4; ++j) {
        const int n0 = (wid * 4 + j) * 16;
        v8f c0 = {}, c1 = {}, c2 = {}, c3 = {};
        for (int k0 = 0; k0 < IN; k0 += 32) {
            v16h b  = load_bfrag(W3xh, IN, n0, k0, lane);
            v16h a0 = load_afrag(xA, XA_STRIDE, 0,  k0, lane);
            v16h a1 = load_afrag(xA, XA_STRIDE, 16, k0, lane);
            v16h a2 = load_afrag(xA, XA_STRIDE, 32, k0, lane);
            v16h a3 = load_afrag(xA, XA_STRIDE, 48, k0, lane);
            c0 = wmma16(a0, b, c0); c1 = wmma16(a1, b, c1);
            c2 = wmma16(a2, b, c2); c3 = wmma16(a3, b, c3);
        }
        {   // retrieved part (K = 32)
            v16h b  = load_bfrag(W3rh, 32, n0, 0, lane);
            v16h a0 = load_afrag(rbuf, 32, 0,  0, lane);
            v16h a1 = load_afrag(rbuf, 32, 16, 0, lane);
            v16h a2 = load_afrag(rbuf, 32, 32, 0, lane);
            v16h a3 = load_afrag(rbuf, 32, 48, 0, lane);
            c0 = wmma16(a0, b, c0); c1 = wmma16(a1, b, c1);
            c2 = wmma16(a2, b, c2); c3 = wmma16(a3, b, c3);
        }
        const int n = n0 + (lane & 15);
        const float bias = b3[n];
        store_relu(hA, H_STRIDE, 0,  n, c0, bias, lh);
        store_relu(hA, H_STRIDE, 16, n, c1, bias, lh);
        store_relu(hA, H_STRIDE, 32, n, c2, bias, lh);
        store_relu(hA, H_STRIDE, 48, n, c3, bias, lh);
    }
    __syncthreads();

    // ---- GEMM4: h2 = relu(h @ W4^T + b4) -> hB (aliases dead x tile region) ----
    for (int j = 0; j < 4; ++j) {
        const int n0 = (wid * 4 + j) * 16;
        v8f c0 = {}, c1 = {}, c2 = {}, c3 = {};
        for (int k0 = 0; k0 < HID; k0 += 32) {
            v16h b  = load_bfrag(W4h, HID, n0, k0, lane);
            v16h a0 = load_afrag(hA, H_STRIDE, 0,  k0, lane);
            v16h a1 = load_afrag(hA, H_STRIDE, 16, k0, lane);
            v16h a2 = load_afrag(hA, H_STRIDE, 32, k0, lane);
            v16h a3 = load_afrag(hA, H_STRIDE, 48, k0, lane);
            c0 = wmma16(a0, b, c0); c1 = wmma16(a1, b, c1);
            c2 = wmma16(a2, b, c2); c3 = wmma16(a3, b, c3);
        }
        const int n = n0 + (lane & 15);
        const float bias = b4[n];
        store_relu(hB, H_STRIDE, 0,  n, c0, bias, lh);
        store_relu(hB, H_STRIDE, 16, n, c1, bias, lh);
        store_relu(hB, H_STRIDE, 32, n, c2, bias, lh);
        store_relu(hB, H_STRIDE, 48, n, c3, bias, lh);
    }
    __syncthreads();

    // ---- GEMM5: out = h2 @ W5p^T + b5 -> global ----
    {
        const int mt = wid & 3;
        const int m0 = mt * 16;
        const int nh = wid >> 2;
        const int n0 = nh * 16;
        v8f c = {};
        for (int k0 = 0; k0 < HID; k0 += 32) {
            v16h a = load_afrag(hB, H_STRIDE, m0, k0, lane);
            v16h b = load_bfrag(W5p, HID, n0, k0, lane);
            c = wmma16(a, b, c);
        }
        const int n = n0 + (lane & 15);
        if (n < OUTD) {
            const float bias = b5[n];
            for (int r = 0; r < 8; ++r) {
                int rr = row0 + m0 + r + (lh ? 0 : 8);
                out[(size_t)rr * OUTD + n] = c[r] + bias;
            }
        }
    }
}

extern "C" void kernel_launch(void* const* d_in, const int* in_sizes, int n_in,
                              void* d_out, int out_size, void* d_ws, size_t ws_size,
                              hipStream_t stream) {
    const float* x   = (const float*)d_in[0];
    const float* W1  = (const float*)d_in[1];
    const float* b1  = (const float*)d_in[2];
    const float* W2  = (const float*)d_in[3];
    const float* b2  = (const float*)d_in[4];
    const float* W3  = (const float*)d_in[5];
    const float* b3  = (const float*)d_in[6];
    const float* W4  = (const float*)d_in[7];
    const float* b4  = (const float*)d_in[8];
    const float* W5  = (const float*)d_in[9];
    const float* b5  = (const float*)d_in[10];
    const float* mem = (const float*)d_in[11];
    char* ws   = (char*)d_ws;
    float* out = (float*)d_out;

    hipLaunchKernelGGL(hop_prep, dim3(256), dim3(256), 0, stream,
                       W1, W2, W3, W4, W5, mem, ws);
    hipLaunchKernelGGL(hop_main, dim3(BATCH / TM), dim3(NTHREADS), 0, stream,
                       x, b1, b2, b3, b4, b5, (const char*)ws, out);
}
